// PCDAlign_22084721836161
// MI455X (gfx1250) — compile-verified, matrix operand order checked by reading, not run on hardware
//
#include <hip/hip_runtime.h>
#include <math.h>

typedef __bf16 bf16;
typedef __attribute__((ext_vector_type(16))) __bf16 v16bf;
typedef __attribute__((ext_vector_type(8)))  __bf16 v8bf;
typedef __attribute__((ext_vector_type(8)))  float   v8f;

#define NFC 64
#define DGR 8
#define NB  4

// ---------------------------------------------------------------------------
// Weight repack: fp32 [O][Cin][3][3] -> bf16 [Opad][Ktot], K = tap*Cin + c
// ---------------------------------------------------------------------------
__global__ void pack_weights_kernel(const float* __restrict__ w, bf16* __restrict__ out,
                                    int O, int Opad, int Cin) {
  int Ktot = Cin * 9;
  int idx = blockIdx.x * 256 + threadIdx.x;
  if (idx >= Opad * Ktot) return;
  int k = idx % Ktot;
  int o = idx / Ktot;
  int tap = k / Cin;
  int c = k - tap * Cin;
  float v = (o < O) ? w[(o * Cin + c) * 9 + tap] : 0.f;
  out[idx] = (bf16)v;
}

// ---------------------------------------------------------------------------
// Bilinear 2x upsample (half-pixel centers, edge clamp) with output scale
// ---------------------------------------------------------------------------
__global__ void up2_kernel(const float* __restrict__ in, float* __restrict__ out,
                           int C, int H, int W, float scale) {
  int HO = 2 * H, WO = 2 * W;
  int total = NB * C * HO * WO;
  int idx = blockIdx.x * 256 + threadIdx.x;
  if (idx >= total) return;
  int wo = idx % WO; int t = idx / WO;
  int ho = t % HO;   t /= HO;
  int c  = t % C;    int b = t / C;
  float sy = (ho + 0.5f) * 0.5f - 0.5f;
  float sx = (wo + 0.5f) * 0.5f - 0.5f;
  float fy = floorf(sy), fx = floorf(sx);
  int y0 = (int)fy, x0 = (int)fx;
  float wy = sy - fy, wx = sx - fx;
  int y0c = min(max(y0, 0), H - 1), y1c = min(max(y0 + 1, 0), H - 1);
  int x0c = min(max(x0, 0), W - 1), x1c = min(max(x0 + 1, 0), W - 1);
  const float* p = in + (b * C + c) * H * W;
  float v = (1.f - wy) * ((1.f - wx) * p[y0c * W + x0c] + wx * p[y0c * W + x1c])
          +         wy * ((1.f - wx) * p[y1c * W + x0c] + wx * p[y1c * W + x1c]);
  out[idx] = v * scale;
}

// ---------------------------------------------------------------------------
// 3x3 conv (pad 1) as implicit GEMM on WMMA bf16, fully templated:
// SZ = H = W (32/64/128) -> all offsets compile-time, e*HW folds into IOFFSET.
// Block: 256 threads = 8 waves; wave = 16-pixel N-tile, OTILES 16-row M-tiles.
// Double-buffered LDS weight tile staged with b128 load/store pairs; one
// barrier per K-chunk, next-chunk loads issued before current WMMAs.
// ---------------------------------------------------------------------------
template <int OTILES, int SZ, int C1, int C2, int OC>
__global__ __launch_bounds__(256)
void conv3x3_wmma_kernel(const float* __restrict__ x1,
                         const float* __restrict__ x2,
                         const bf16* __restrict__ Wgt, const float* __restrict__ bias,
                         float* __restrict__ out, int act) {
  constexpr int Cin  = C1 + C2;       // 64 or 128 (powers of two)
  constexpr int Ktot = Cin * 9;
  constexpr int HW   = SZ * SZ;
  constexpr int NCH  = Ktot / 32;
  __shared__ bf16 sW[2][OTILES * 16 * 40];   // 40-elem pitch: 16B aligned, bank spread

  const int tid  = threadIdx.x;
  const int lane = tid & 31;
  const int wave = tid >> 5;
  const int half = lane >> 4;
  const int nidx = lane & 15;

  const int p = blockIdx.x * 128 + wave * 16 + nidx;
  const int b = p / HW;           // HW is a power of two -> shift
  const int r = p & (HW - 1);
  const int h = r / SZ;
  const int w = r & (SZ - 1);

  // Stage one 32-wide K-chunk of weights: 8 bf16 (16B) per segment,
  // one global_load_b128 + one ds_store_b128 each.
  auto stage = [&](int buf, int kb) {
    constexpr int SEGS = OTILES * 16 * 4;   // 4 segments per weight row
#pragma unroll
    for (int s0 = 0; s0 < SEGS; s0 += 256) {
      int s = s0 + tid;
      if (SEGS % 256 == 0 || s < SEGS) {
        int o = s >> 2, seg = (s & 3) * 8;
        v8bf v = *(const v8bf*)(Wgt + o * Ktot + kb + seg);
        *(v8bf*)(&sW[buf][o * 40 + seg]) = v;
      }
    }
  };

  v8f acc[OTILES];
#pragma unroll
  for (int t = 0; t < OTILES; ++t) acc[t] = (v8f){0.f, 0.f, 0.f, 0.f, 0.f, 0.f, 0.f, 0.f};

  stage(0, 0);
  __syncthreads();

  for (int ch = 0; ch < NCH; ++ch) {
    const int kb  = ch * 32;
    const int buf = ch & 1;
    if (ch + 1 < NCH) {
      stage(buf ^ 1, kb + 32);                      // prefetch next weight chunk
      __builtin_prefetch(Wgt + kb + 64, 0, 1);      // and hint the one after
    }

    const int tap   = kb / Cin;                     // chunk stays in one tap
    const int cbase = kb & (Cin - 1);
    const int dy = tap / 3 - 1, dx = tap - (tap / 3) * 3 - 1;
    const int hh = h + dy, ww = w + dx;
    const bool inb = (hh >= 0) & (hh < SZ) & (ww >= 0) & (ww < SZ);

    v16bf bfrag = {};
    if (inb) {
      const int pix = hh * SZ + ww;
      const int c0 = cbase + 8 * half;              // 8-aligned: never straddles concat
      const int c1 = c0 + 16;
      const float* p0; int o0;
      const float* p1; int o1;
      if (C2 == 0 || c0 < C1) { p0 = x1; o0 = (b * C1 + c0) * HW + pix; }
      else                    { p0 = x2; o0 = (b * C2 + (c0 - C1)) * HW + pix; }
      if (C2 == 0 || c1 < C1) { p1 = x1; o1 = (b * C1 + c1) * HW + pix; }
      else                    { p1 = x2; o1 = (b * C2 + (c1 - C1)) * HW + pix; }
#pragma unroll
      for (int e = 0; e < 8; ++e) {                 // e*HW folds into IOFFSET
        bfrag[e]     = (bf16)p0[o0 + e * HW];
        bfrag[e + 8] = (bf16)p1[o1 + e * HW];
      }
    }

#pragma unroll
    for (int t = 0; t < OTILES; ++t) {
      const bf16* wrow = &sW[buf][(t * 16 + nidx) * 40 + half * 8];
      v8bf lo = *(const v8bf*)(wrow);
      v8bf hi = *(const v8bf*)(wrow + 16);
      v16bf afrag;
#pragma unroll
      for (int e = 0; e < 8; ++e) { afrag[e] = lo[e]; afrag[e + 8] = hi[e]; }
      acc[t] = __builtin_amdgcn_wmma_f32_16x16x32_bf16(false, afrag, false, bfrag,
                                                       (short)0, acc[t], false, false);
    }
    __syncthreads();
  }

  const int hw = h * SZ + w;
#pragma unroll
  for (int t = 0; t < OTILES; ++t) {
#pragma unroll
    for (int rr = 0; rr < 8; ++rr) {
      const int m = t * 16 + half * 8 + rr;
      if (m < OC) {
        float v = acc[t][rr] + bias[m];
        if (act) v = (v >= 0.f) ? v : 0.1f * v;
        out[(b * OC + m) * HW + hw] = v;
      }
    }
  }
}

// ---------------------------------------------------------------------------
// Modulated deformable 3x3 conv fused with WMMA GEMM (SZ templated).
// om: [B, 3*DG*9, H, W] raw conv output (o1 | o2 | mask-logits).
// K = tap*64 + c; each 32-chunk = one tap; each lane touches exactly 2
// deformable groups -> offsets/mask/corner-voffsets hoisted per group, all
// 64 gathers become base+imm loads off 8 precomputed voffsets.
// ---------------------------------------------------------------------------
template <int SZ>
__global__ __launch_bounds__(256)
void dcn_wmma_kernel(const float* __restrict__ x, const float* __restrict__ om,
                     const bf16* __restrict__ Wgt, const float* __restrict__ bias,
                     float* __restrict__ out, int act) {
  constexpr int OT = 4;
  constexpr int Ktot = NFC * 9;
  constexpr int HW = SZ * SZ;
  constexpr int NCH = Ktot / 32;
  __shared__ bf16 sW[2][OT * 16 * 40];

  const int tid  = threadIdx.x;
  const int lane = tid & 31;
  const int wave = tid >> 5;
  const int half = lane >> 4;
  const int nidx = lane & 15;

  const int p = blockIdx.x * 128 + wave * 16 + nidx;
  const int b = p / HW;
  const int r = p & (HW - 1);
  const int h = r / SZ;
  const int w = r & (SZ - 1);

  auto stage = [&](int buf, int kb) {
    constexpr int SEGS = OT * 16 * 4;
#pragma unroll
    for (int s0 = 0; s0 < SEGS; s0 += 256) {
      int s = s0 + tid;
      if (SEGS % 256 == 0 || s < SEGS) {
        int o = s >> 2, seg = (s & 3) * 8;
        v8bf v = *(const v8bf*)(Wgt + o * Ktot + kb + seg);
        *(v8bf*)(&sW[buf][o * 40 + seg]) = v;
      }
    }
  };

  v8f acc[OT];
#pragma unroll
  for (int t = 0; t < OT; ++t) acc[t] = (v8f){0.f, 0.f, 0.f, 0.f, 0.f, 0.f, 0.f, 0.f};

  stage(0, 0);
  __syncthreads();

  for (int ch = 0; ch < NCH; ++ch) {
    const int kb  = ch * 32;
    const int buf = ch & 1;
    if (ch + 1 < NCH) stage(buf ^ 1, kb + 32);

    const int tap   = kb >> 6;
    const int cbase = kb & 63;
    const int dy = tap / 3 - 1, dx = tap - (tap / 3) * 3 - 1;

    float gw[2][4];
    int   gv[2][4];     // per-group gather voffsets (element index into x)
#pragma unroll
    for (int g = 0; g < 2; ++g) {
      const int dg  = (cbase >> 3) + half + g * 2;
      const int och = dg * 9 + tap;
      const int ombase = b * (3 * DGR * 9) * HW + h * SZ + w;
      float oy = om[ombase + och * HW];
      float ox = om[ombase + (DGR * 9 + och) * HW];
      float mk = om[ombase + (2 * DGR * 9 + och) * HW];
      float ms = 1.f / (1.f + __expf(-mk));
      float py = oy + (float)(dy + h);
      float px = ox + (float)(dx + w);
      float fy = floorf(py), fx = floorf(px);
      int y0 = (int)fy, x0 = (int)fx;
      float wy = py - fy, wx = px - fx;
      int y1 = y0 + 1, x1i = x0 + 1;
      bool vy0 = (y0 >= 0) & (y0 < SZ),  vy1 = (y1 >= 0) & (y1 < SZ);
      bool vx0 = (x0 >= 0) & (x0 < SZ),  vx1 = (x1i >= 0) & (x1i < SZ);
      int y0c = min(max(y0, 0), SZ - 1),  y1c = min(max(y1, 0), SZ - 1);
      int x0c = min(max(x0, 0), SZ - 1),  x1c = min(max(x1i, 0), SZ - 1);
      gw[g][0] = (vy0 & vx0) ? (1.f - wy) * (1.f - wx) * ms : 0.f;
      gw[g][1] = (vy0 & vx1) ? (1.f - wy) * wx * ms : 0.f;
      gw[g][2] = (vy1 & vx0) ? wy * (1.f - wx) * ms : 0.f;
      gw[g][3] = (vy1 & vx1) ? wy * wx * ms : 0.f;
      const int cg = cbase + 8 * half + 16 * g;
      const int xb = (b * NFC + cg) * HW;
      gv[g][0] = xb + y0c * SZ + x0c; gv[g][1] = xb + y0c * SZ + x1c;
      gv[g][2] = xb + y1c * SZ + x0c; gv[g][3] = xb + y1c * SZ + x1c;
    }

    v16bf bfrag;
#pragma unroll
    for (int e = 0; e < 8; ++e) {                   // e*HW folds into IOFFSET
      float s0 = gw[0][0] * x[gv[0][0] + e * HW] + gw[0][1] * x[gv[0][1] + e * HW]
               + gw[0][2] * x[gv[0][2] + e * HW] + gw[0][3] * x[gv[0][3] + e * HW];
      float s1 = gw[1][0] * x[gv[1][0] + e * HW] + gw[1][1] * x[gv[1][1] + e * HW]
               + gw[1][2] * x[gv[1][2] + e * HW] + gw[1][3] * x[gv[1][3] + e * HW];
      bfrag[e]     = (bf16)s0;
      bfrag[e + 8] = (bf16)s1;
    }

#pragma unroll
    for (int t = 0; t < OT; ++t) {
      const bf16* wrow = &sW[buf][(t * 16 + nidx) * 40 + half * 8];
      v8bf lo = *(const v8bf*)(wrow);
      v8bf hi = *(const v8bf*)(wrow + 16);
      v16bf afrag;
#pragma unroll
      for (int e = 0; e < 8; ++e) { afrag[e] = lo[e]; afrag[e + 8] = hi[e]; }
      acc[t] = __builtin_amdgcn_wmma_f32_16x16x32_bf16(false, afrag, false, bfrag,
                                                       (short)0, acc[t], false, false);
    }
    __syncthreads();
  }

  const int hw = h * SZ + w;
#pragma unroll
  for (int t = 0; t < OT; ++t) {
#pragma unroll
    for (int rr = 0; rr < 8; ++rr) {
      const int m = t * 16 + half * 8 + rr;
      float v = acc[t][rr] + bias[m];
      if (act) v = (v >= 0.f) ? v : 0.1f * v;
      out[(b * NFC + m) * HW + hw] = v;
    }
  }
}

// ---------------------------------------------------------------------------
// Host orchestration
// ---------------------------------------------------------------------------
extern "C" void kernel_launch(void* const* d_in, const int* in_sizes, int n_in,
                              void* d_out, int out_size, void* d_ws, size_t ws_size,
                              hipStream_t stream) {
  (void)in_sizes; (void)n_in; (void)out_size; (void)ws_size;

  const float* fea1_l1 = (const float*)d_in[0];
  const float* fea1_l2 = (const float*)d_in[1];
  const float* fea1_l3 = (const float*)d_in[2];
  const float* fea2_l1 = (const float*)d_in[3];
  const float* fea2_l2 = (const float*)d_in[4];
  const float* fea2_l3 = (const float*)d_in[5];
  const float* w128 = (const float*)d_in[6];
  const float* b128 = (const float*)d_in[7];
  const float* w64  = (const float*)d_in[8];
  const float* b64  = (const float*)d_in[9];
  const float* om_w = (const float*)d_in[10];
  const float* om_b = (const float*)d_in[11];
  const float* dcn_w = (const float*)d_in[12];
  const float* dcn_b = (const float*)d_in[13];

  // ---- workspace bump allocator ----
  char* ws = (char*)d_ws;
  size_t off = 0;
  auto alloc = [&](size_t bytes) -> void* {
    void* r = ws + off;
    off += (bytes + 255) & ~(size_t)255;
    return r;
  };

  bf16* Wp128 = (bf16*)alloc((size_t)8 * 64 * 1152 * 2);
  bf16* Wp64  = (bf16*)alloc((size_t)4 * 64 * 576 * 2);
  bf16* Wpom  = (bf16*)alloc((size_t)4 * 224 * 576 * 2);
  bf16* Wpdcn = (bf16*)alloc((size_t)4 * 64 * 576 * 2);

  const size_t F3 = (size_t)NB * NFC * 32 * 32;
  const size_t F2 = (size_t)NB * NFC * 64 * 64;
  const size_t F1 = (size_t)NB * NFC * 128 * 128;
  float* A3 = (float*)alloc(F3 * 4);
  float* B3 = (float*)alloc(F3 * 4);
  float* OM3 = (float*)alloc((size_t)NB * 216 * 32 * 32 * 4);
  float* A2 = (float*)alloc(F2 * 4);
  float* B2 = (float*)alloc(F2 * 4);
  float* C2 = (float*)alloc(F2 * 4);
  float* D2 = (float*)alloc(F2 * 4);
  float* OM2 = (float*)alloc((size_t)NB * 216 * 64 * 64 * 4);
  float* A1 = (float*)alloc(F1 * 4);
  float* B1 = (float*)alloc(F1 * 4);
  float* C1b = (float*)alloc(F1 * 4);
  float* D1 = (float*)alloc(F1 * 4);
  float* OM1 = (float*)alloc((size_t)NB * 216 * 128 * 128 * 4);

  // ---- pack weights (bf16, tap-major K) ----
  auto pack = [&](const float* src, bf16* dst, int O, int Opad, int Cin) {
    int total = Opad * Cin * 9;
    pack_weights_kernel<<<(total + 255) / 256, 256, 0, stream>>>(src, dst, O, Opad, Cin);
  };
  for (int m = 0; m < 8; ++m) pack(w128 + (size_t)m * 64 * 128 * 9, Wp128 + (size_t)m * 64 * 1152, 64, 64, 128);
  for (int m = 0; m < 4; ++m) pack(w64  + (size_t)m * 64 * 64 * 9,  Wp64  + (size_t)m * 64 * 576,  64, 64, 64);
  for (int m = 0; m < 4; ++m) pack(om_w + (size_t)m * 216 * 64 * 9, Wpom  + (size_t)m * 224 * 576, 216, 224, 64);
  for (int m = 0; m < 4; ++m) pack(dcn_w + (size_t)m * 64 * 64 * 9, Wpdcn + (size_t)m * 64 * 576,  64, 64, 64);

  auto W128 = [&](int m) { return Wp128 + (size_t)m * 64 * 1152; };
  auto W64p = [&](int m) { return Wp64 + (size_t)m * 64 * 576; };
  auto up2 = [&](const float* in, float* outp, int Hin, int Win, float scale) {
    int total = NB * NFC * 4 * Hin * Win;
    up2_kernel<<<(total + 255) / 256, 256, 0, stream>>>(in, outp, NFC, Hin, Win, scale);
  };

#define GRID(SZ) (NB * (SZ) * (SZ) / 128)

  // ---- L3 (32x32) ----
  conv3x3_wmma_kernel<4, 32, 64, 64, 64><<<GRID(32), 256, 0, stream>>>(
      fea1_l3, fea2_l3, W128(0), b128 + 0 * 64, A3, 1);                        // t3a
  conv3x3_wmma_kernel<4, 32, 64, 0, 64><<<GRID(32), 256, 0, stream>>>(
      A3, A3, W64p(0), b64 + 0 * 64, B3, 1);                                   // L3_off
  conv3x3_wmma_kernel<14, 32, 64, 0, 216><<<GRID(32), 256, 0, stream>>>(
      B3, B3, Wpom + 0 * 224 * 576, om_b + 0 * 216, OM3, 0);
  dcn_wmma_kernel<32><<<GRID(32), 256, 0, stream>>>(
      fea1_l3, OM3, Wpdcn + 0 * 64 * 576, dcn_b + 0 * 64, A3, 1);              // L3_fea

  // ---- L2 (64x64) ----
  conv3x3_wmma_kernel<4, 64, 64, 64, 64><<<GRID(64), 256, 0, stream>>>(
      fea1_l2, fea2_l2, W128(1), b128 + 1 * 64, A2, 1);                        // t2a
  up2(B3, B2, 32, 32, 2.0f);                                                   // up(L3_off)*2
  conv3x3_wmma_kernel<4, 64, 64, 64, 64><<<GRID(64), 256, 0, stream>>>(
      A2, B2, W128(2), b128 + 2 * 64, C2, 1);                                  // t2b
  conv3x3_wmma_kernel<4, 64, 64, 0, 64><<<GRID(64), 256, 0, stream>>>(
      C2, C2, W64p(1), b64 + 1 * 64, A2, 1);                                   // L2_off
  conv3x3_wmma_kernel<14, 64, 64, 0, 216><<<GRID(64), 256, 0, stream>>>(
      A2, A2, Wpom + 1 * 224 * 576, om_b + 1 * 216, OM2, 0);
  dcn_wmma_kernel<64><<<GRID(64), 256, 0, stream>>>(
      fea1_l2, OM2, Wpdcn + 1 * 64 * 576, dcn_b + 1 * 64, B2, 0);              // t2c
  up2(A3, C2, 32, 32, 1.0f);                                                   // up(L3_fea)
  conv3x3_wmma_kernel<4, 64, 64, 64, 64><<<GRID(64), 256, 0, stream>>>(
      B2, C2, W128(3), b128 + 3 * 64, D2, 1);                                  // L2_fea

  // ---- L1 (128x128) ----
  conv3x3_wmma_kernel<4, 128, 64, 64, 64><<<GRID(128), 256, 0, stream>>>(
      fea1_l1, fea2_l1, W128(4), b128 + 4 * 64, A1, 1);                        // t1a
  up2(A2, B1, 64, 64, 2.0f);                                                   // up(L2_off)*2
  conv3x3_wmma_kernel<4, 128, 64, 64, 64><<<GRID(128), 256, 0, stream>>>(
      A1, B1, W128(5), b128 + 5 * 64, C1b, 1);                                 // t1b
  conv3x3_wmma_kernel<4, 128, 64, 0, 64><<<GRID(128), 256, 0, stream>>>(
      C1b, C1b, W64p(2), b64 + 2 * 64, A1, 1);                                 // L1_off
  conv3x3_wmma_kernel<14, 128, 64, 0, 216><<<GRID(128), 256, 0, stream>>>(
      A1, A1, Wpom + 2 * 224 * 576, om_b + 2 * 216, OM1, 0);
  dcn_wmma_kernel<128><<<GRID(128), 256, 0, stream>>>(
      fea1_l1, OM1, Wpdcn + 2 * 64 * 576, dcn_b + 2 * 64, B1, 0);              // t1c
  up2(D2, D1, 64, 64, 1.0f);                                                   // up(L2_fea)
  conv3x3_wmma_kernel<4, 128, 64, 64, 64><<<GRID(128), 256, 0, stream>>>(
      B1, D1, W128(6), b128 + 6 * 64, C1b, 0);                                 // L1_fea

  // ---- Cascade ----
  conv3x3_wmma_kernel<4, 128, 64, 64, 64><<<GRID(128), 256, 0, stream>>>(
      C1b, fea2_l1, W128(7), b128 + 7 * 64, A1, 1);
  conv3x3_wmma_kernel<4, 128, 64, 0, 64><<<GRID(128), 256, 0, stream>>>(
      A1, A1, W64p(3), b64 + 3 * 64, B1, 1);
  conv3x3_wmma_kernel<14, 128, 64, 0, 216><<<GRID(128), 256, 0, stream>>>(
      B1, B1, Wpom + 3 * 224 * 576, om_b + 3 * 216, OM1, 0);
  dcn_wmma_kernel<128><<<GRID(128), 256, 0, stream>>>(
      C1b, OM1, Wpdcn + 3 * 64 * 576, dcn_b + 3 * 64, (float*)d_out, 1);
#undef GRID
}
